// MixerModel_2199023255673
// MI455X (gfx1250) — compile-verified, matrix operand order checked by reading, not run on hardware
//
#include <hip/hip_runtime.h>
#include <hip/hip_bf16.h>
#include <math.h>

// ---------------------------------------------------------------------------
// Mamba mixer (4 blocks) for gfx1250.
// GEMMs: double-buffered TDM (tensor_load_to_lds) pipeline stages f16 A/B
// panels into LDS; 8 waves per block feed v_wmma_f32_16x16x32_f16 from
// batched ds_load_b128 fragments.
// ---------------------------------------------------------------------------

typedef __attribute__((ext_vector_type(16))) _Float16 v16h;
typedef __attribute__((ext_vector_type(8)))  _Float16 v8h;
typedef __attribute__((ext_vector_type(8)))  float    v8f;
typedef __attribute__((ext_vector_type(4)))  unsigned int v4u;
typedef __attribute__((ext_vector_type(8)))  int      v8i;
typedef __attribute__((ext_vector_type(4)))  int      v4i;

#define D_MODEL 512
#define D_INNER 1024
#define D_STATE 16
#define D_CONV  4
#define DT_RANK 32
#define NB      4
#define BSZ     2
#define LSEQ    2048
#define ROWS    (BSZ * LSEQ)   // 4096

#define HAS_TDM __has_builtin(__builtin_amdgcn_tensor_load_to_lds)

// ------------------- TDM: 2-D f16 tile load (global -> LDS) ----------------
// D# per CDNA5 ISA ch.8: group0 = {flags, lds_addr, global_addr, type=2},
// group1 = {data_size=2B, tensor dims, tile dims, dim0 stride}.
#if HAS_TDM
__device__ __forceinline__ void tdm_load_2d_f16(
        unsigned int lds_off, const _Float16* gptr,
        unsigned int tile_k, unsigned int tile_rows, unsigned int row_stride) {
    unsigned long long ga = (unsigned long long)(uintptr_t)gptr;
    v4u g0;
    g0[0] = 1u;                                             // count=1 (valid), no gather
    g0[1] = lds_off;                                        // lds_addr [63:32]
    g0[2] = (unsigned int)ga;                               // global_addr[31:0]
    g0[3] = (unsigned int)((ga >> 32) & 0x01ffffffu) | (2u << 30); // addr[56:32], type=2
    v8i g1;
    g1[0] = (int)(1u << 16);                                // data_size=1 -> 2 bytes
    g1[1] = (int)((tile_k & 0xffffu) << 16);                // tensor_dim0[15:0]
    g1[2] = (int)((tile_k >> 16) | ((tile_rows & 0xffffu) << 16)); // td0 hi | td1 lo
    g1[3] = (int)((tile_rows >> 16) | ((tile_k & 0xffffu) << 16)); // td1 hi | tile_dim0
    g1[4] = (int)(tile_rows & 0xffffu);                     // tile_dim1 (tile_dim2=0)
    g1[5] = (int)row_stride;                                // tensor_dim0_stride[31:0]
    g1[6] = 0;
    g1[7] = 0;
    v4i z4 = {0, 0, 0, 0};
#if defined(__clang_major__) && (__clang_major__ >= 23)
    v8i z8 = {0, 0, 0, 0, 0, 0, 0, 0};
    __builtin_amdgcn_tensor_load_to_lds(g0, g1, z4, z4, z8, 0);
#else
    __builtin_amdgcn_tensor_load_to_lds(g0, g1, z4, z4, 0);
#endif
}
#endif

// --------------------------- LayerNorm -------------------------------------
__global__ void ln_kernel(const float* __restrict__ x, const float* __restrict__ w,
                          const float* __restrict__ b, float* __restrict__ outf,
                          _Float16* __restrict__ outh, int D) {
    __shared__ float s1[256];
    __shared__ float s2[256];
    const int row = blockIdx.x;
    const int tid = threadIdx.x;
    const float* xr = x + (size_t)row * D;
    float sum = 0.f, sq = 0.f;
    for (int i = tid; i < D; i += 256) { float v = xr[i]; sum += v; sq += v * v; }
    s1[tid] = sum; s2[tid] = sq;
    __syncthreads();
    for (int off = 128; off > 0; off >>= 1) {
        if (tid < off) { s1[tid] += s1[tid + off]; s2[tid] += s2[tid + off]; }
        __syncthreads();
    }
    const float mu   = s1[0] / (float)D;
    const float var  = s2[0] / (float)D - mu * mu;
    const float rstd = rsqrtf(var + 1e-5f);
    for (int i = tid; i < D; i += 256) {
        const float v = (xr[i] - mu) * rstd * w[i] + b[i];
        if (outf) outf[(size_t)row * D + i] = v;
        if (outh) outh[(size_t)row * D + i] = (_Float16)v;
    }
}

// --------------------- f32 -> f16 weight conversion -------------------------
__global__ void cvt_f16_kernel(const float* __restrict__ src,
                               _Float16* __restrict__ dst, int n) {
    const int i = blockIdx.x * 256 + threadIdx.x;
    if (i < n) dst[i] = (_Float16)src[i];
}

// --------------------------- TDM + WMMA GEMM --------------------------------
// C[m,n] = sum_k A[m,k] * Bw[n,k]  (f16 operands, K-contiguous, strides lda/ldb)
// Block: 256 threads (8 waves) -> 128 x (16*NSUB) tile.  Double-buffered TDM
// pipeline: wave 0 keeps the NEXT K-step's A/B panels in flight while all
// waves compute the CURRENT step from LDS (batched ds_load fragments).
// epi==1: v = softplus(v + bias[n]);  flipL>0: row remap b*L+(L-1-t);
// C16 != null: also store f16 copy.
template <int NSUB>
__global__ __launch_bounds__(256) void gemm_tdm(
        const _Float16* __restrict__ A, const _Float16* __restrict__ Bw,
        float* __restrict__ C, _Float16* __restrict__ C16,
        int K, int lda, int ldb, int ldc,
        const float* __restrict__ bias, int epi, int flipL) {
    __shared__ __align__(16) _Float16 sA[2 * 128 * 32];
    __shared__ __align__(16) _Float16 sB[2 * 16 * NSUB * 32];

    const int tid  = threadIdx.x;
    const int wave = tid >> 5;
    const int lane = tid & 31;
    const int nlo  = lane & 15;
    const int kgrp = (lane >> 4) * 8;      // ISA 16-bit fragment k-group (0 or 8)
    const int bm = blockIdx.y;
    const int bn = blockIdx.x;
    const int nsteps = K >> 5;

    const _Float16* Abase = A  + (size_t)(bm * 128) * lda;
    const _Float16* Bbase = Bw + (size_t)(bn * 16 * NSUB) * ldb;

    v8f acc[NSUB];
#pragma unroll
    for (int j = 0; j < NSUB; ++j) acc[j] = (v8f){0.f,0.f,0.f,0.f,0.f,0.f,0.f,0.f};

#if HAS_TDM
    const unsigned int ldsA = (unsigned int)(uintptr_t)(&sA[0]);
    const unsigned int ldsB = (unsigned int)(uintptr_t)(&sB[0]);
    // ---- pipeline prologue: steps 0 and 1 in flight
    if (tid < 32) {
        tdm_load_2d_f16(ldsA, Abase, 32u, 128u, (unsigned int)lda);
        tdm_load_2d_f16(ldsB, Bbase, 32u, (unsigned int)(16 * NSUB), (unsigned int)ldb);
        if (nsteps > 1) {
            tdm_load_2d_f16(ldsA + 128 * 32 * 2, Abase + 32, 32u, 128u, (unsigned int)lda);
            tdm_load_2d_f16(ldsB + 16 * NSUB * 32 * 2, Bbase + 32, 32u,
                            (unsigned int)(16 * NSUB), (unsigned int)ldb);
        }
    }
#endif

    for (int s = 0; s < nsteps; ++s) {
        const int buf = s & 1;
#if HAS_TDM
        if (tid < 32) {            // current step's panels have landed
            if (s + 1 < nsteps) __builtin_amdgcn_s_wait_tensorcnt(2);
            else                __builtin_amdgcn_s_wait_tensorcnt(0);
        }
#else
        {   // cooperative fallback copy of step s
            const int kk = s * 32;
            for (int e = tid; e < 128 * 4; e += 256) {
                const int r = e >> 2, c = (e & 3) * 8;
                *(v8h*)(sA + buf * 128 * 32 + r * 32 + c) =
                    *(const v8h*)(Abase + (size_t)r * lda + kk + c);
            }
            for (int e = tid; e < 16 * NSUB * 4; e += 256) {
                const int r = e >> 2, c = (e & 3) * 8;
                *(v8h*)(sB + buf * 16 * NSUB * 32 + r * 32 + c) =
                    *(const v8h*)(Bbase + (size_t)r * ldb + kk + c);
            }
        }
#endif
        __syncthreads();           // panels visible to all waves

        // ---- batched fragment loads (one ds clause), then the wmma chain
        const _Float16* sAb = sA + buf * 128 * 32;
        const _Float16* sBb = sB + buf * 16 * NSUB * 32;
        const _Float16* ar = sAb + (wave * 16 + nlo) * 32 + kgrp;
        v8h a0 = *(const v8h*)(ar);
        v8h a1 = *(const v8h*)(ar + 16);
        v16h af = __builtin_shufflevector(a0, a1, 0,1,2,3,4,5,6,7,8,9,10,11,12,13,14,15);

        v16h bf[NSUB];
#pragma unroll
        for (int j = 0; j < NSUB; ++j) {
            const _Float16* br = sBb + (j * 16 + nlo) * 32 + kgrp;
            v8h b0 = *(const v8h*)(br);
            v8h b1 = *(const v8h*)(br + 16);
            bf[j] = __builtin_shufflevector(b0, b1, 0,1,2,3,4,5,6,7,8,9,10,11,12,13,14,15);
        }
#pragma unroll
        for (int j = 0; j < NSUB; ++j) {
            acc[j] = __builtin_amdgcn_wmma_f32_16x16x32_f16(
                false, af, false, bf[j], (short)0, acc[j], false, false);
        }

        __syncthreads();           // all reads of buf done before refill
#if HAS_TDM
        if (tid < 32 && s + 2 < nsteps) {   // keep next-next step in flight
            const int kk2 = (s + 2) * 32;
            tdm_load_2d_f16(ldsA + buf * 128 * 32 * 2, Abase + kk2, 32u, 128u,
                            (unsigned int)lda);
            tdm_load_2d_f16(ldsB + buf * 16 * NSUB * 32 * 2, Bbase + kk2, 32u,
                            (unsigned int)(16 * NSUB), (unsigned int)ldb);
        }
#endif
    }

    // ---- epilogue (C/D layout: VGPR r -> M = r + 8*(lane>>4), N = lane&15)
    const int halfsel = lane >> 4;
#pragma unroll
    for (int j = 0; j < NSUB; ++j) {
        const int n = bn * (16 * NSUB) + j * 16 + nlo;
#pragma unroll
        for (int r = 0; r < 8; ++r) {
            const int m = bm * 128 + wave * 16 + halfsel * 8 + r;
            float v = acc[j][r];
            if (epi == 1) {
                v += bias[n];
                v = (v > 20.f) ? v : log1pf(__expf(v));
            }
            int mrow = m;
            if (flipL > 0) {
                const int bb = m / flipL;
                const int tt = m - bb * flipL;
                mrow = bb * flipL + (flipL - 1 - tt);
            }
            C[(size_t)mrow * ldc + n] = v;
            if (C16) C16[(size_t)mrow * ldc + n] = (_Float16)v;
        }
    }
}

// ------------------- depthwise causal conv (k=4) + SiLU ---------------------
__global__ void conv_silu_kernel(const float* __restrict__ xz,
                                 const float* __restrict__ cw,
                                 const float* __restrict__ cb,
                                 float* __restrict__ xic,
                                 _Float16* __restrict__ xic16) {
    const int idx = blockIdx.x * blockDim.x + threadIdx.x;   // B*L*D_INNER
    const int d = idx & (D_INNER - 1);
    const int t = (idx >> 10) & (LSEQ - 1);
    const int b = idx >> 21;
    float acc = cb[d];
#pragma unroll
    for (int k = 0; k < D_CONV; ++k) {
        const int tt = t - (D_CONV - 1) + k;
        if (tt >= 0)
            acc += xz[((size_t)(b * LSEQ + tt)) * (2 * D_INNER) + d] * cw[d * D_CONV + k];
    }
    const float s = acc / (1.f + __expf(-acc));              // silu
    xic[idx] = s;
    xic16[idx] = (_Float16)s;
}

// ------------------- selective scan: lane = (b, d, n) -----------------------
__global__ void scan_kernel(const float* __restrict__ dt,
                            const float* __restrict__ xdbl,
                            const float* __restrict__ xic,
                            const float* __restrict__ xz,
                            const float* __restrict__ A_log,
                            const float* __restrict__ D_p,
                            _Float16* __restrict__ y16) {
    const int tid = blockIdx.x * blockDim.x + threadIdx.x;   // B*D_INNER*16
    const int n = tid & 15;
    const int d = (tid >> 4) & (D_INNER - 1);
    const int b = tid >> 14;
    const float Ac = -__expf(A_log[d * D_STATE + n]);
    const float Dv = D_p[d];
    float h = 0.f;
    for (int t = 0; t < LSEQ; ++t) {
        const size_t base = (size_t)(b * LSEQ + t);
        const float dtv = dt[base * D_INNER + d];
        const float xiv = xic[base * D_INNER + d];
        const float Bv  = xdbl[base * 64 + DT_RANK + n];
        const float Cv  = xdbl[base * 64 + DT_RANK + D_STATE + n];
        h = __expf(dtv * Ac) * h + dtv * Bv * xiv;
        float p = h * Cv;
        p += __shfl_xor(p, 1, 16);
        p += __shfl_xor(p, 2, 16);
        p += __shfl_xor(p, 4, 16);
        p += __shfl_xor(p, 8, 16);
        if (n == 0) {
            const float zv = xz[base * (2 * D_INNER) + D_INNER + d];
            const float yv = p + Dv * xiv;
            y16[base * D_INNER + d] = (_Float16)(yv * (zv / (1.f + __expf(-zv))));
        }
    }
}

// ---------------------------------------------------------------------------
extern "C" void kernel_launch(void* const* d_in, const int* in_sizes, int n_in,
                              void* d_out, int out_size, void* d_ws, size_t ws_size,
                              hipStream_t stream) {
    const float* x      = (const float*)d_in[0];
    const float* ln_w   = (const float*)d_in[1];
    const float* ln_b   = (const float*)d_in[2];
    const float* W_in   = (const float*)d_in[3];
    const float* conv_w = (const float*)d_in[4];
    const float* conv_b = (const float*)d_in[5];
    const float* W_x    = (const float*)d_in[6];
    const float* W_dt   = (const float*)d_in[7];
    const float* b_dt   = (const float*)d_in[8];
    const float* A_log  = (const float*)d_in[9];
    const float* D_p    = (const float*)d_in[10];
    const float* W_out  = (const float*)d_in[11];
    const float* nf_w   = (const float*)d_in[12];
    const float* nf_b   = (const float*)d_in[13];

    char* w = (char*)d_ws;
    _Float16* xn16   = (_Float16*)w;  w += (size_t)ROWS * D_MODEL   * 2;  // 4 MB
    float*    xz     = (float*)w;     w += (size_t)ROWS * 2*D_INNER * 4;  // 32 MB
    float*    xic    = (float*)w;     w += (size_t)ROWS * D_INNER   * 4;  // 16 MB
    _Float16* xic16  = (_Float16*)w;  w += (size_t)ROWS * D_INNER   * 2;  // 8 MB
    float*    xdbl   = (float*)w;     w += (size_t)ROWS * 64        * 4;  // 1 MB
    _Float16* xdbl16 = (_Float16*)w;  w += (size_t)ROWS * 64        * 2;  // 0.5 MB
    float*    dtb    = (float*)w;     w += (size_t)ROWS * D_INNER   * 4;  // 16 MB
    _Float16* yb16   = (_Float16*)w;  w += (size_t)ROWS * D_INNER   * 2;  // 8 MB
    float*    h0     = (float*)w;     w += (size_t)ROWS * D_MODEL   * 4;  // 8 MB
    _Float16* Win16  = (_Float16*)w;  w += (size_t)NB * 2*D_INNER * D_MODEL * 2;
    _Float16* Wx16   = (_Float16*)w;  w += (size_t)NB * 64 * D_INNER * 2;
    _Float16* Wdt16  = (_Float16*)w;  w += (size_t)NB * D_INNER * DT_RANK * 2;
    _Float16* Wout16 = (_Float16*)w;  w += (size_t)NB * D_MODEL * D_INNER * 2;

    // weights -> f16 (once per launch)
    {
        int n1 = NB * 2 * D_INNER * D_MODEL;
        int n2 = NB * 64 * D_INNER;
        int n3 = NB * D_INNER * DT_RANK;
        int n4 = NB * D_MODEL * D_INNER;
        cvt_f16_kernel<<<(n1 + 255) / 256, 256, 0, stream>>>(W_in,  Win16,  n1);
        cvt_f16_kernel<<<(n2 + 255) / 256, 256, 0, stream>>>(W_x,   Wx16,   n2);
        cvt_f16_kernel<<<(n3 + 255) / 256, 256, 0, stream>>>(W_dt,  Wdt16,  n3);
        cvt_f16_kernel<<<(n4 + 255) / 256, 256, 0, stream>>>(W_out, Wout16, n4);
    }

    const float* cur = x;
    for (int i = 0; i < NB; ++i) {
        // 1) layernorm -> f16 activations
        ln_kernel<<<ROWS, 256, 0, stream>>>(cur, ln_w + i*D_MODEL, ln_b + i*D_MODEL,
                                            nullptr, xn16, D_MODEL);
        // 2) xz = xn @ W_in^T      (4096 x 2048, K=512)
        gemm_tdm<8><<<dim3((2*D_INNER)/128, ROWS/128), 256, 0, stream>>>(
            xn16, Win16 + (size_t)i*2*D_INNER*D_MODEL, xz, nullptr,
            D_MODEL, D_MODEL, D_MODEL, 2*D_INNER, nullptr, 0, 0);
        // 3) depthwise conv + silu   (f32 for scan, f16 for next GEMM)
        conv_silu_kernel<<<(ROWS*D_INNER)/256, 256, 0, stream>>>(
            xz, conv_w + (size_t)i*D_INNER*D_CONV, conv_b + (size_t)i*D_INNER, xic, xic16);
        // 4) x_dbl = xic @ W_x^T   (4096 x 64, K=1024) -> dual f32/f16 store
        gemm_tdm<4><<<dim3(1, ROWS/128), 256, 0, stream>>>(
            xic16, Wx16 + (size_t)i*64*D_INNER, xdbl, xdbl16,
            D_INNER, D_INNER, D_INNER, 64, nullptr, 0, 0);
        // 5) dt = softplus(x_dbl[:, :32] @ W_dt^T + b_dt)  (4096 x 1024, K=32)
        gemm_tdm<8><<<dim3(D_INNER/128, ROWS/128), 256, 0, stream>>>(
            xdbl16, Wdt16 + (size_t)i*D_INNER*DT_RANK, dtb, nullptr,
            DT_RANK, 64, DT_RANK, D_INNER, b_dt + (size_t)i*D_INNER, 1, 0);
        // 6) selective scan + D*xi + silu(z) gating -> f16 y
        scan_kernel<<<(BSZ*D_INNER*D_STATE)/256, 256, 0, stream>>>(
            dtb, xdbl, xic, xz, A_log + (size_t)i*D_INNER*D_STATE,
            D_p + (size_t)i*D_INNER, yb16);
        // 7) h = flipL( y @ W_out^T )   (4096 x 512, K=1024)
        gemm_tdm<8><<<dim3(D_MODEL/128, ROWS/128), 256, 0, stream>>>(
            yb16, Wout16 + (size_t)i*D_MODEL*D_INNER, h0, nullptr,
            D_INNER, D_INNER, D_INNER, D_MODEL, nullptr, 0, LSEQ);
        cur = h0;
    }
    // final layernorm -> d_out (fp32)
    ln_kernel<<<ROWS, 256, 0, stream>>>(cur, nf_w, nf_b, (float*)d_out, nullptr, D_MODEL);
}
